// MSAModule_23124103922381
// MI455X (gfx1250) — matrix operand order repaired and checked by reading, not compiled, not run on hardware
//
#include <hip/hip_runtime.h>
#include <math.h>

// ---------------------------------------------------------------------------
// MI455X (gfx1250) implementation of the MSA / Evoformer reference.
// All dense math runs through v_wmma_f32_16x16x32_bf16 (f32 accumulate).
// LDS tiles are staged so fragments load as ds_load_b128 pairs (no scalar
// ds_load_u16 gathers); all register arrays are statically indexed (no movrel).
// ---------------------------------------------------------------------------

typedef __bf16 bf16_t;
typedef __attribute__((ext_vector_type(16))) __bf16 v16bf;
typedef __attribute__((ext_vector_type(8)))  __bf16 v8bf;
typedef __attribute__((ext_vector_type(8)))  float  v8f;

#define EF_BIAS 1
#define EF_SIG  2
#define EF_MUL  4
#define EF_RES  8
#define EF_TRES 16   // residual/store at transposed (j,i) location, N=256

static __device__ __forceinline__ bf16_t f2bf(float f) {
  unsigned u = __builtin_bit_cast(unsigned, f);
  unsigned r = (u + 0x7fffu + ((u >> 16) & 1u)) >> 16;   // RNE f32 -> bf16
  unsigned short h = (unsigned short)r;
  return __builtin_bit_cast(bf16_t, h);
}
static __device__ __forceinline__ float sigmoidf_(float x) {
  return 1.0f / (1.0f + expf(-x));
}

// CDNA5 WMMA bf16 fragment layouts (wave32), per CDNA5 ISA:
//  A (16x32):  m = lane&15 ; elems 0-7 -> K = (lane/16)*8 + 0..7,
//              elems 8-15 -> K = 16 + (lane/16)*8 + 0..7  (two contiguous runs)
//  B (32x16):  n = lane&15 ; same K mapping as A (two contiguous runs)
//  C/D (16x16 f32, 8 VGPR): row = r + (lane/16)*8, col = lane&15
static __device__ __forceinline__ int akidx(int i, int lane) {
  return ((i >> 3) << 4) + ((lane >> 4) << 3) + (i & 7);
}

// ---- vectorized LDS fragment loads (rows K-contiguous, stride%8==0) --------
static __device__ __forceinline__ v16bf ldsAfrag(const bf16_t* p, int stride) {
  int lane = threadIdx.x & 31;
  const bf16_t* row = p + (lane & 15) * stride + ((lane >> 4) << 3);
  v8bf c0 = *(const v8bf*)row;          // K run 0..7   (+lanehalf*8)
  v8bf c1 = *(const v8bf*)(row + 16);   // K run 16..23 (+lanehalf*8)
  return __builtin_shufflevector(c0, c1, 0, 1, 2, 3, 4, 5, 6, 7,
                                 8, 9, 10, 11, 12, 13, 14, 15);
}
static __device__ __forceinline__ v16bf ldsBfrag(const bf16_t* p, int stride) {
  // p points at (n=0, k=0) of a transposed [n][k] tile
  int lane = threadIdx.x & 31;
  const bf16_t* row = p + (lane & 15) * stride + ((lane >> 4) << 3);
  v8bf c0 = *(const v8bf*)row;
  v8bf c1 = *(const v8bf*)(row + 16);
  return __builtin_shufflevector(c0, c1, 0, 1, 2, 3, 4, 5, 6, 7,
                                 8, 9, 10, 11, 12, 13, 14, 15);
}

// ---- scalar global-gather fragment loads (for strided einsum operands) -----
static __device__ __forceinline__ v16bf lafrag_f32(const float* p, long sm, long sk) {
  int lane = threadIdx.x & 31;
  const float* pm = p + (long)(lane & 15) * sm;
  v16bf a;
#pragma unroll
  for (int i = 0; i < 16; ++i) a[i] = f2bf(pm[(long)akidx(i, lane) * sk]);
  return a;
}
static __device__ __forceinline__ v16bf lbfrag_f32(const float* p, long sk, long sn) {
  int lane = threadIdx.x & 31;
  const float* pn = p + (long)(lane & 15) * sn;
  v16bf b;
#pragma unroll
  for (int i = 0; i < 16; ++i) b[i] = f2bf(pn[(long)akidx(i, lane) * sk]);
  return b;
}
static __device__ __forceinline__ v8f wmma_bf16(v16bf a, v16bf b, v8f c) {
  return __builtin_amdgcn_wmma_f32_16x16x32_bf16(false, a, false, b, (short)0, c,
                                                 false, false);
}
static __device__ __forceinline__ v8f vz8() {
  v8f z;
#pragma unroll
  for (int e = 0; e < 8; ++e) z[e] = 0.f;
  return z;
}

// ---------------- LayerNorm (one wave per row, D<=128) ----------------------
__global__ void k_ln(const float* __restrict__ x, const float* __restrict__ g,
                     const float* __restrict__ b, const float* __restrict__ gate,
                     float* __restrict__ out, int rows, int D) {
  int row = blockIdx.x * 4 + (threadIdx.x >> 5);
  if (row >= rows) return;
  int lane = threadIdx.x & 31;
  const float* xr = x + (long)row * D;
  int nchunk = D >> 5;
  float xv[4];
  float s = 0.f, s2 = 0.f;
#pragma unroll
  for (int t = 0; t < 4; ++t)
    if (t < nchunk) {
      float v = xr[lane + t * 32];
      xv[t] = v; s += v; s2 += v * v;
    }
  for (int o = 16; o; o >>= 1) { s += __shfl_xor(s, o); s2 += __shfl_xor(s2, o); }
  float inv = 1.f / (float)D;
  float mu = s * inv;
  float var = s2 * inv - mu * mu;
  float rs = rsqrtf(var + 1e-5f);
  float* orow = out + (long)row * D;
  const float* gr = gate ? gate + (long)row * D : nullptr;
#pragma unroll
  for (int t = 0; t < 4; ++t)
    if (t < nchunk) {
      int c = lane + t * 32;
      float v = (xv[t] - mu) * rs * g[c] + b[c];
      if (gr) v *= gr[c];
      orow[c] = v;
    }
}

// ---------------- Main WMMA GEMM: C[M,Nt] = epi(A[M,K] @ W[K,Nt]) -----------
// 128 threads = 4 waves; block tile 64 x (NT*16); LDS-staged bf16 operands.
// W tile staged TRANSPOSED [col][k] so B-fragments are ds_load_b128 pairs.
template <int NT>
__global__ void k_gemm_t(const float* __restrict__ A, const float* __restrict__ W,
                         const float* __restrict__ bias, const float* __restrict__ mulbuf,
                         float* __restrict__ C, int M, int K, int Nt, int flags) {
  __shared__ bf16_t lA[64][40];
  __shared__ bf16_t lWt[NT * 16][40];
  const int wcols = NT * 16;
  int tid = threadIdx.x;
  int w = tid >> 5, lane = tid & 31;
  int m0 = blockIdx.x * 64;
  int n0 = blockIdx.y * wcols;
  v8f acc[NT];
#pragma unroll
  for (int t = 0; t < NT; ++t) acc[t] = vz8();
  for (int k0 = 0; k0 < K; k0 += 32) {
    for (int e = tid; e < 2048; e += 128) {
      int r = e >> 5, c = e & 31;
      lA[r][c] = f2bf(A[(long)(m0 + r) * K + k0 + c]);
    }
    for (int e = tid; e < 32 * wcols; e += 128) {
      int c = e % wcols, r = e / wcols;   // coalesced read of W row k0+r
      lWt[c][r] = f2bf(W[(long)(k0 + r) * Nt + n0 + c]);
    }
    if (k0 + 32 < K)  // gfx1250 global_prefetch_b8 for the next A tile
      __builtin_prefetch(&A[(long)(m0 + (tid >> 1)) * K + k0 + 32], 0, 1);
    __syncthreads();
    v16bf af = ldsAfrag(&lA[w * 16][0], 40);
#pragma unroll
    for (int t = 0; t < NT; ++t) {
      v16bf bf = ldsBfrag(&lWt[t * 16][0], 40);
      acc[t] = wmma_bf16(af, bf, acc[t]);
    }
    __syncthreads();
  }
  int rowb = m0 + w * 16 + ((lane >> 4) << 3);
#pragma unroll
  for (int t = 0; t < NT; ++t) {
    int col = n0 + t * 16 + (lane & 15);
#pragma unroll
    for (int r = 0; r < 8; ++r) {
      int row = rowb + r;
      float v = acc[t][r];
      if (flags & EF_BIAS) v += bias[col];
      if (flags & EF_SIG)  v = sigmoidf_(v);
      if (flags & EF_MUL)  v *= mulbuf[(long)row * Nt + col];
      long drow = row;
      if (flags & EF_TRES) drow = (long)(row & 255) * 256 + (row >> 8);
      long di = drow * (long)Nt + col;
      if (flags & EF_RES)  v += C[di];
      C[di] = v;
    }
  }
}

// Dual GEMM: C = (A@W1) * f(A@W2), f = sigmoid if sig2 (gated projections).
__global__ void k_gemm_dual(const float* __restrict__ A, const float* __restrict__ W1,
                            const float* __restrict__ W2, float* __restrict__ C,
                            int M, int K, int Nt, int sig2) {
  __shared__ bf16_t lA[64][40];
  __shared__ bf16_t lW1t[64][40];
  __shared__ bf16_t lW2t[64][40];
  int tid = threadIdx.x;
  int w = tid >> 5, lane = tid & 31;
  int m0 = blockIdx.x * 64;
  int n0 = blockIdx.y * 64;
  v8f acc1[4], acc2[4];
#pragma unroll
  for (int t = 0; t < 4; ++t) { acc1[t] = vz8(); acc2[t] = vz8(); }
  for (int k0 = 0; k0 < K; k0 += 32) {
    for (int e = tid; e < 2048; e += 128) {
      int r = e >> 5, c = e & 31;
      lA[r][c] = f2bf(A[(long)(m0 + r) * K + k0 + c]);
    }
    for (int e = tid; e < 2048; e += 128) {
      int c = e & 63, r = e >> 6;
      lW1t[c][r] = f2bf(W1[(long)(k0 + r) * Nt + n0 + c]);
      lW2t[c][r] = f2bf(W2[(long)(k0 + r) * Nt + n0 + c]);
    }
    __syncthreads();
    v16bf af = ldsAfrag(&lA[w * 16][0], 40);
#pragma unroll
    for (int t = 0; t < 4; ++t) {
      v16bf b1 = ldsBfrag(&lW1t[t * 16][0], 40);
      v16bf b2 = ldsBfrag(&lW2t[t * 16][0], 40);
      acc1[t] = wmma_bf16(af, b1, acc1[t]);
      acc2[t] = wmma_bf16(af, b2, acc2[t]);
    }
    __syncthreads();
  }
  int rowb = m0 + w * 16 + ((lane >> 4) << 3);
#pragma unroll
  for (int t = 0; t < 4; ++t) {
    int col = n0 + t * 16 + (lane & 15);
#pragma unroll
    for (int r = 0; r < 8; ++r) {
      int row = rowb + r;
      float v2 = acc2[t][r];
      if (sig2) v2 = sigmoidf_(v2);
      C[(long)row * Nt + col] = acc1[t][r] * v2;
    }
  }
}

// ---------------- Outer product mean, fused with w3 projection --------------
// block = (i, 16 j's): phase1 WMMA outer product over s (K=64) -> LDS bf16
// tile [16 j][1024 ef]; phase2 WMMA x w3 (K=1024) -> pair residual add.
__global__ void k_opm_fused(const float* __restrict__ aw, const float* __restrict__ bw,
                            const float* __restrict__ w3, const float* __restrict__ b3,
                            float* __restrict__ pair) {
  __shared__ bf16_t Ot[16][1040];
  int i = blockIdx.x;
  int j0 = blockIdx.y * 16;
  int w = threadIdx.x >> 5, lane = threadIdx.x & 31;  // 8 waves
#pragma unroll
  for (int jh = 0; jh < 2; ++jh) {
    int jj = 2 * w + jh;
    int j = j0 + jj;
#pragma unroll
    for (int et = 0; et < 2; ++et)
#pragma unroll
      for (int ft = 0; ft < 2; ++ft) {
        v8f acc = vz8();
#pragma unroll
        for (int kk = 0; kk < 64; kk += 32) {
          // A(e,s) = a[s,i,e]  (sm=1, sk=N*32=8192) ; B(s,f) = b[s,j,f]
          v16bf af = lafrag_f32(aw + (long)kk * 8192 + i * 32 + et * 16, 1, 8192);
          v16bf bf = lbfrag_f32(bw + (long)kk * 8192 + j * 32 + ft * 16, 8192, 1);
          acc = wmma_bf16(af, bf, acc);
        }
        int mb = (lane >> 4) << 3, n = lane & 15;
#pragma unroll
        for (int r = 0; r < 8; ++r)
          Ot[jj][(et * 16 + mb + r) * 32 + ft * 16 + n] = f2bf(acc[r] * (1.f / 64.f));
      }
  }
  __syncthreads();
  v8f acc = vz8();  // rows = 16 jj, cols = w*16..w*16+15 of 128
  for (int kk = 0; kk < 1024; kk += 32) {
    v16bf af = ldsAfrag(&Ot[0][kk], 1040);
    v16bf bf = lbfrag_f32(w3 + (long)kk * 128 + w * 16, 128, 1);
    acc = wmma_bf16(af, bf, acc);
  }
  int mb = (lane >> 4) << 3;
  int col = w * 16 + (lane & 15);
#pragma unroll
  for (int r = 0; r < 8; ++r) {
    int jj = mb + r;
    long di = ((long)i * 256 + j0 + jj) * 128 + col;
    pair[di] += acc[r] + b3[col];
  }
}

// ---------------- PWA: out[s,i,h,d] = (sum_j w[i,j,h] v[s,j,h,d]) * g -------
__global__ void k_pwa_attn(const float* __restrict__ w8, const float* __restrict__ vg,
                           float* __restrict__ outb) {
  int i0 = blockIdx.x * 16;
  int h = blockIdx.y;
  int s = blockIdx.z;
  int w = threadIdx.x >> 5, lane = threadIdx.x & 31;  // 2 waves: d-tiles
  v8f acc = vz8();
  for (int j0 = 0; j0 < 256; j0 += 32) {
    v16bf af = lafrag_f32(w8 + ((long)i0 * 256 + j0) * 8 + h, 2048, 8);
    v16bf bf = lbfrag_f32(vg + ((long)s * 256 + j0) * 512 + h * 32 + w * 16, 512, 1);
    acc = wmma_bf16(af, bf, acc);
  }
  int mb = (lane >> 4) << 3;
  int d = w * 16 + (lane & 15);
#pragma unroll
  for (int r = 0; r < 8; ++r) {
    int i = i0 + mb + r;
    float gv = sigmoidf_(vg[((long)s * 256 + i) * 512 + 256 + h * 32 + d]);
    outb[((long)s * 256 + i) * 256 + h * 32 + d] = acc[r] * gv;
  }
}

// ---------------- Triangle multiply einsum (per-channel 256x256x256) --------
// mode 0 (outgoing): out[i,j,d] = sum_k L[i,k,d] R[j,k,d]
// mode 1 (incoming): out[i,j,d] = sum_k L[k,j,d] R[k,i,d]
__global__ void k_trimul_einsum(const float* __restrict__ L, const float* __restrict__ R,
                                float* __restrict__ O, int mode) {
  int i0 = blockIdx.x * 16, j0 = blockIdx.y * 16;
  int w = threadIdx.x >> 5, lane = threadIdx.x & 31;  // 4 waves, d each
  int d = blockIdx.z * 4 + w;
  v8f acc = vz8();
  for (int k0 = 0; k0 < 256; k0 += 32) {
    const float *pA, *pB; long sAm, sAk, sBk, sBn;
    if (mode == 0) {
      pA = L + ((long)i0 * 256 + k0) * 128 + d; sAm = 32768; sAk = 128;
      pB = R + ((long)j0 * 256 + k0) * 128 + d; sBk = 128;   sBn = 32768;
    } else {
      pA = R + ((long)k0 * 256 + i0) * 128 + d; sAm = 128;   sAk = 32768;
      pB = L + ((long)k0 * 256 + j0) * 128 + d; sBk = 32768; sBn = 128;
    }
    acc = wmma_bf16(lafrag_f32(pA, sAm, sAk), lbfrag_f32(pB, sBk, sBn), acc);
  }
  int mb = (lane >> 4) << 3, n = lane & 15;
#pragma unroll
  for (int r = 0; r < 8; ++r)
    O[((long)(i0 + mb + r) * 256 + j0 + n) * 128 + d] = acc[r];
}

// ---------------- Fused triangle attention per (h, i, 16-row j tile) --------
__global__ void k_tri_attn(const float* __restrict__ q, const float* __restrict__ kt,
                           const float* __restrict__ vt, const float* __restrict__ gt,
                           const float* __restrict__ bias4, float* __restrict__ outb) {
  __shared__ bf16_t Vt[32][264];    // transposed [d][kcol] -> B-frag b128 loads
  __shared__ bf16_t Ql[16][40];
  __shared__ float  Sl[16][257];
  __shared__ bf16_t Pl[16][264];
  int jb = blockIdx.x, i = blockIdx.y, h = blockIdx.z;
  int tid = threadIdx.x, w = tid >> 5, lane = tid & 31;  // 8 waves
  for (int e = tid; e < 8192; e += 256) {
    int kc = e >> 5, d = e & 31;
    Vt[d][kc] = f2bf(vt[((long)i * 256 + kc) * 128 + h * 32 + d]);
  }
  if (tid < 512) {
    int jl = tid >> 5, d = tid & 31;
    Ql[jl][d] = f2bf(q[((long)i * 256 + jb * 16 + jl) * 128 + h * 32 + d]);
    Ql[jl + 8][d] = f2bf(q[((long)i * 256 + jb * 16 + jl + 8) * 128 + h * 32 + d]);
  }
  __syncthreads();
  // scores: 16(j) x 256(k), K = d = 32 (single WMMA per tile)
  v16bf af = ldsAfrag(&Ql[0][0], 40);
#pragma unroll
  for (int kk = 0; kk < 2; ++kk) {
    int ktile = w + kk * 8;
    v16bf bf = lbfrag_f32(kt + ((long)i * 256 + ktile * 16) * 128 + h * 32, 1, 128);
    v8f acc = vz8();
    acc = wmma_bf16(af, bf, acc);
    int mb = (lane >> 4) << 3, n = lane & 15;
#pragma unroll
    for (int r = 0; r < 8; ++r) {
      int m = mb + r, kc = ktile * 16 + n;
      Sl[m][kc] = acc[r] * 0.1767766952966369f +
                  bias4[((long)(jb * 16 + m) * 256 + kc) * 4 + h];
    }
  }
  __syncthreads();
#pragma unroll
  for (int rr = 0; rr < 2; ++rr) {   // softmax rows (2 per wave)
    int m = w * 2 + rr;
    float v[8], mx = -1e30f;
#pragma unroll
    for (int t = 0; t < 8; ++t) { v[t] = Sl[m][lane + t * 32]; mx = fmaxf(mx, v[t]); }
    for (int o = 16; o; o >>= 1) mx = fmaxf(mx, __shfl_xor(mx, o));
    float sum = 0.f;
#pragma unroll
    for (int t = 0; t < 8; ++t) { v[t] = expf(v[t] - mx); sum += v[t]; }
    for (int o = 16; o; o >>= 1) sum += __shfl_xor(sum, o);
    float is = 1.f / sum;
#pragma unroll
    for (int t = 0; t < 8; ++t) Pl[m][lane + t * 32] = f2bf(v[t] * is);
  }
  __syncthreads();
  if (w < 2) {  // out = P @ V  (16 x 32, K = 256), gated
    v8f acc = vz8();
    for (int k0 = 0; k0 < 256; k0 += 32) {
      v16bf pa = ldsAfrag(&Pl[0][k0], 264);
      v16bf vb = ldsBfrag(&Vt[w * 16][k0], 264);
      acc = wmma_bf16(pa, vb, acc);
    }
    int mb = (lane >> 4) << 3;
    int d = w * 16 + (lane & 15);
#pragma unroll
    for (int r = 0; r < 8; ++r) {
      int j = jb * 16 + mb + r;
      long di = ((long)i * 256 + j) * 128 + h * 32 + d;
      outb[di] = acc[r] * gt[di];
    }
  }
}

// ---------------- Small helpers (VALU) --------------------------------------
__global__ void k_small_gemm(const float* __restrict__ A, const float* __restrict__ W,
                             float* __restrict__ C, int M, int K, int Nt) {
  long t = (long)blockIdx.x * 256 + threadIdx.x;
  if (t >= (long)M * Nt) return;
  int r = (int)(t / Nt), c = (int)(t % Nt);
  float s = 0.f;
  for (int k = 0; k < K; ++k) s += A[(long)r * K + k] * W[k * Nt + c];
  C[t] = s;
}
__global__ void k_softmax8(const float* __restrict__ in, float* __restrict__ out) {
  int row = blockIdx.x * 4 + (threadIdx.x >> 5);   // (i,h) rows: 2048
  int lane = threadIdx.x & 31;
  int i = row >> 3, h = row & 7;
  float v[8], mx = -1e30f;
#pragma unroll
  for (int t = 0; t < 8; ++t) {
    int j = lane + t * 32;
    v[t] = in[((long)i * 256 + j) * 8 + h];
    mx = fmaxf(mx, v[t]);
  }
  for (int o = 16; o; o >>= 1) mx = fmaxf(mx, __shfl_xor(mx, o));
  float s = 0.f;
#pragma unroll
  for (int t = 0; t < 8; ++t) { v[t] = expf(v[t] - mx); s += v[t]; }
  for (int o = 16; o; o >>= 1) s += __shfl_xor(s, o);
  float is = 1.f / s;
#pragma unroll
  for (int t = 0; t < 8; ++t)
    out[((long)i * 256 + lane + t * 32) * 8 + h] = v[t] * is;
}
__global__ void k_addb(float* __restrict__ msa, const float* __restrict__ sp) {
  int t = blockIdx.x * 256 + threadIdx.x;          // 16384*64
  msa[t] += sp[t & 16383];
}
__global__ void k_transpose128(const float* __restrict__ in, float* __restrict__ out) {
  long t = (long)blockIdx.x * 256 + threadIdx.x;   // 65536*128
  int c = (int)(t & 127);
  long row = t >> 7;
  int i = (int)(row >> 8), j = (int)(row & 255);
  out[(((long)j << 8) + i) * 128 + c] = in[t];
}

// ---------------------------------------------------------------------------
extern "C" void kernel_launch(void* const* d_in, const int* in_sizes, int n_in,
                              void* d_out, int out_size, void* d_ws, size_t ws_size,
                              hipStream_t stream) {
  (void)in_sizes; (void)n_in; (void)out_size; (void)ws_size;
  const float* single = (const float*)d_in[0];   // (256,384)
  const float* pairin = (const float*)d_in[1];   // (256,256,128)
  const float* msain  = (const float*)d_in[2];   // (64,256,64)
  int idx = 3;
  auto nx = [&]() { return (const float*)d_in[idx++]; };

  // workspace layout (floats) — needs ~456 MB
  float* wsf = (float*)d_ws;
  const long PE = 65536L * 128;
  float* pair  = wsf; wsf += PE;
  float* pairT = wsf; wsf += PE;
  float* xn    = wsf; wsf += PE;
  float* t0    = wsf; wsf += PE;
  float* t1    = wsf; wsf += PE;
  float* t2    = wsf; wsf += PE;
  float* t3    = wsf; wsf += PE;
  float* t4    = wsf; wsf += PE;
  float* big   = wsf; wsf += 65536L * 512;   // ptrans h12 / pwa vg (aliased)
  float* msa   = wsf; wsf += 16384L * 64;
  float* mln   = wsf; wsf += 16384L * 64;
  float* sproj = wsf; wsf += 256L * 64;
  float* aw    = wsf; wsf += 16384L * 32;
  float* bw    = wsf; wsf += 16384L * 32;
  float* h12m  = wsf; wsf += 16384L * 256;
  float* pwab  = wsf; wsf += 16384L * 256;
  float* bias8 = wsf; wsf += 65536L * 8;
  float* w8b   = wsf; wsf += 65536L * 8;
  float* bias4 = wsf; wsf += 65536L * 4;
  float* vg    = big;

  auto gemm = [&](const float* A, const float* W, float* C, int M, int K, int Nt,
                  int flags, const float* bias, const float* mul) {
    if (Nt >= 64) {
      dim3 grid(M / 64, Nt / 64);
      k_gemm_t<4><<<grid, 128, 0, stream>>>(A, W, bias, mul, C, M, K, Nt, flags);
    } else {
      dim3 grid(M / 64, Nt / 32);
      k_gemm_t<2><<<grid, 128, 0, stream>>>(A, W, bias, mul, C, M, K, Nt, flags);
    }
  };
  auto gemmdual = [&](const float* A, const float* W1, const float* W2, float* C,
                      int M, int K, int Nt, int sig2) {
    dim3 grid(M / 64, Nt / 64);
    k_gemm_dual<<<grid, 128, 0, stream>>>(A, W1, W2, C, M, K, Nt, sig2);
  };
  auto ln = [&](const float* x, const float* g, const float* b, const float* gate,
                float* o, int rows, int D) {
    k_ln<<<dim3(rows / 4), 128, 0, stream>>>(x, g, b, gate, o, rows, D);
  };

  // ---- init: pair copy, msa = msa@Wi + broadcast(single@Ws) ----------------
  hipMemcpyAsync(pair, pairin, PE * sizeof(float), hipMemcpyDeviceToDevice, stream);
  const float* msa_init_w = nx();
  const float* single_w   = nx();
  gemm(msain, msa_init_w, msa, 16384, 64, 64, 0, nullptr, nullptr);
  gemm(single, single_w, sproj, 256, 384, 64, 0, nullptr, nullptr);
  k_addb<<<dim3(4096), 256, 0, stream>>>(msa, sproj);

  for (int l = 0; l < 2; ++l) {
    const float* opm[6];  for (int t = 0; t < 6;  ++t) opm[t]  = nx();
    const float* pwa[7];  for (int t = 0; t < 7;  ++t) pwa[t]  = nx();
    const float* mt[5];   for (int t = 0; t < 5;  ++t) mt[t]   = nx();
    const float* tmo[10]; for (int t = 0; t < 10; ++t) tmo[t]  = nx();
    const float* tmi[10]; for (int t = 0; t < 10; ++t) tmi[t]  = nx();
    const float* tas[8];  for (int t = 0; t < 8;  ++t) tas[t]  = nx();
    const float* tae[8];  for (int t = 0; t < 8;  ++t) tae[t]  = nx();
    const float* pt[5];   for (int t = 0; t < 5;  ++t) pt[t]   = nx();

    // ---- outer product mean -> pair +=
    ln(msa, opm[0], opm[1], nullptr, mln, 16384, 64);
    gemm(mln, opm[2], aw, 16384, 64, 32, 0, nullptr, nullptr);
    gemm(mln, opm[3], bw, 16384, 64, 32, 0, nullptr, nullptr);
    k_opm_fused<<<dim3(256, 16), 256, 0, stream>>>(aw, bw, opm[4], opm[5], pair);

    // ---- pair-weighted averaging -> msa +=
    ln(msa, pwa[0], pwa[1], nullptr, mln, 16384, 64);
    gemm(mln, pwa[2], vg, 16384, 64, 512, 0, nullptr, nullptr);
    ln(pair, pwa[3], pwa[4], nullptr, xn, 65536, 128);
    k_small_gemm<<<dim3(2048), 256, 0, stream>>>(xn, pwa[5], bias8, 65536, 128, 8);
    k_softmax8<<<dim3(512), 128, 0, stream>>>(bias8, w8b);
    k_pwa_attn<<<dim3(16, 8, 64), 64, 0, stream>>>(w8b, vg, pwab);
    gemm(pwab, pwa[6], msa, 16384, 256, 64, EF_RES, nullptr, nullptr);

    // ---- msa transition -> msa +=
    ln(msa, mt[0], mt[1], nullptr, mln, 16384, 64);
    gemmdual(mln, mt[2], mt[3], h12m, 16384, 64, 256, 0);
    gemm(h12m, mt[4], msa, 16384, 256, 64, EF_RES, nullptr, nullptr);

    // ---- pair block: triangle multiplies
    auto trimult = [&](const float* const* p, int mode) {
      ln(pair, p[0], p[1], nullptr, xn, 65536, 128);
      gemmdual(xn, p[2], p[4], t0, 65536, 128, 128, 1);  // left = wl * sig(wlg)
      gemmdual(xn, p[3], p[5], t1, 65536, 128, 128, 1);  // right
      gemm(xn, p[6], t2, 65536, 128, 128, EF_SIG, nullptr, nullptr);  // og
      k_trimul_einsum<<<dim3(16, 16, 32), 128, 0, stream>>>(t0, t1, t3, mode);
      ln(t3, p[7], p[8], t2, t4, 65536, 128);            // ln(out)*og
      gemm(t4, p[9], pair, 65536, 128, 128, EF_RES, nullptr, nullptr);
    };
    trimult(tmo, 0);
    trimult(tmi, 1);

    // ---- triangle attention (start / end)
    auto triattn = [&](const float* const* p, const float* src, int extra) {
      ln(src, p[0], p[1], nullptr, xn, 65536, 128);
      gemm(xn, p[2], t0, 65536, 128, 128, 0, nullptr, nullptr);       // q
      gemm(xn, p[3], t1, 65536, 128, 128, 0, nullptr, nullptr);       // k
      gemm(xn, p[4], t2, 65536, 128, 128, 0, nullptr, nullptr);       // v
      gemm(xn, p[5], t3, 65536, 128, 128, EF_SIG, nullptr, nullptr);  // g
      k_small_gemm<<<dim3(1024), 256, 0, stream>>>(xn, p[6], bias4, 65536, 128, 4);
      k_tri_attn<<<dim3(16, 256, 4), 256, 0, stream>>>(t0, t1, t2, t3, bias4, t4);
      gemm(t4, p[7], pair, 65536, 128, 128, EF_RES | extra, nullptr, nullptr);
    };
    triattn(tas, pair, 0);
    k_transpose128<<<dim3(32768), 256, 0, stream>>>(pair, pairT);
    triattn(tae, pairT, EF_TRES);   // compute in transposed space, add back transposed

    // ---- pair transition -> pair +=
    ln(pair, pt[0], pt[1], nullptr, xn, 65536, 128);
    gemmdual(xn, pt[2], pt[3], big, 65536, 128, 512, 0);
    gemm(big, pt[4], pair, 65536, 512, 128, EF_RES, nullptr, nullptr);
  }

  hipMemcpyAsync(d_out, pair, PE * sizeof(float), hipMemcpyDeviceToDevice, stream);
}
